// Net_71270687310327
// MI455X (gfx1250) — matrix operand order repaired and checked by reading, not compiled, not run on hardware
//
#include <hip/hip_runtime.h>
#include <hip/hip_bf16.h>

typedef __attribute__((ext_vector_type(16))) __bf16 v16bf;
typedef __attribute__((ext_vector_type(8)))  __bf16 v8bf;
typedef __attribute__((ext_vector_type(2)))  __bf16 v2bf;
typedef __attribute__((ext_vector_type(8)))  float  v8f;

#define BATCH 32
#define MSZ   512
#define VOC   256
#define HID   1024
#define TDB   10
#define DIN   ((VOC + TDB) * MSZ)   // 136192 = 2128 * 64

// ---------------------------------------------------------------------------
// prep: per-batch argmin of decayed surprise, timing bump, stable rank sort,
// scatter memory rows (bf16) + timing bits into A0 [32, DIN].
// One block per batch, 256 threads.
// ---------------------------------------------------------------------------
__global__ __launch_bounds__(256) void prep_kernel(
    const float* __restrict__ x,        // [B, V] one-hot
    const float* __restrict__ memory,   // [B, M, V]
    const float* __restrict__ mtim,     // [B, M]
    const float* __restrict__ msur,     // [B, M]
    __bf16* __restrict__ A0)            // [B, DIN] bf16
{
    __shared__ float tv[MSZ];
    __shared__ int   rk[MSZ];
    __shared__ float redv[256];
    __shared__ int   redi[256];

    const int b = blockIdx.x;
    const int t = threadIdx.x;

    // argmin of msur*DECAY (decay monotone; keep it anyway). First-index ties.
    float v0 = msur[b * MSZ + t]       * 0.9f;
    float v1 = msur[b * MSZ + t + 256] * 0.9f;
    float bv; int bi;
    if (v0 <= v1) { bv = v0; bi = t; } else { bv = v1; bi = t + 256; }
    redv[t] = bv; redi[t] = bi;
    __syncthreads();
    for (int s = 128; s > 0; s >>= 1) {
        if (t < s) {
            float ov = redv[t + s]; int oi = redi[t + s];
            if (ov < redv[t] || (ov == redv[t] && oi < redi[t])) {
                redv[t] = ov; redi[t] = oi;
            }
        }
        __syncthreads();
    }
    const int idx = redi[0];

    // bumped timings, written slot zeroed
    float ta = mtim[b * MSZ + t] + 1.0f;        if (t == idx)       ta = 0.0f;
    float tb = mtim[b * MSZ + t + 256] + 1.0f;  if (t + 256 == idx) tb = 0.0f;
    tv[t] = ta; tv[t + 256] = tb;
    __syncthreads();

    // stable rank of each element under ascending sort (counting sort)
    for (int pass = 0; pass < 2; ++pass) {
        int   m  = t + pass * 256;
        float tm = tv[m];
        int   r  = 0;
        for (int j = 0; j < MSZ; ++j) {
            float tj = tv[j];
            r += (tj < tm) || (tj == tm && j < m);
        }
        rk[m] = r;
    }
    __syncthreads();

    // scatter rows + timing bits (thread t handles vocab column t)
    for (int m = 0; m < MSZ; ++m) {
        const int r = rk[m];
        const float* src = (m == idx) ? (x + (size_t)b * VOC)
                                      : (memory + ((size_t)b * MSZ + m) * VOC);
        A0[(size_t)b * DIN + (size_t)r * VOC + t] = (__bf16)src[t];
        if (t < TDB) {
            unsigned ti = (unsigned)tv[m];
            A0[(size_t)b * DIN + (size_t)MSZ * VOC + r * TDB + t] =
                (__bf16)(float)((ti >> t) & 1u);
        }
    }
}

// ---------------------------------------------------------------------------
// bias init: C[b, n] = bias[n]  (accumulator target for the K-split GEMM)
// ---------------------------------------------------------------------------
__global__ void bias_init(float* __restrict__ C, const float* __restrict__ bias, int N)
{
    int i = blockIdx.x * blockDim.x + threadIdx.x;
    if (i < BATCH * N) C[i] = bias[i % N];
}

// ---------------------------------------------------------------------------
// relu + fp32 -> bf16 for the next layer's A matrix
// ---------------------------------------------------------------------------
__global__ void relu_to_bf16(const float* __restrict__ C, __bf16* __restrict__ A, int n)
{
    int i = blockIdx.x * blockDim.x + threadIdx.x;
    if (i < n) {
        float v = C[i];
        A[i] = (__bf16)(v > 0.0f ? v : 0.0f);
    }
}

// ---------------------------------------------------------------------------
// K-split WMMA GEMM:  C[32,N] += A[32,K]_bf16 @ W[K,N]_f32(->bf16)
//
// Block = 256 threads = 8 waves. Wave w: M-tile (w&1), 16-col N band (w>>1).
// grid.x = N/64 bands, grid.y = K splits; each split runs `kstages` stages of
// K=64 (2 WMMAs per wave per stage).
//
// Software pipeline: next stage's W tile is prefetched into registers with
// NON-TEMPORAL loads (W streams once; keep the L2 for the reused A matrix),
// WMMAs consume the current LDS buffer, prefetched data is stored packed
// (bf16x2 -> ds_store_b32) into the alternate LDS buffer, one barrier per
// stage. LDS layout is [n][k] so the B fragment is two contiguous 16B
// ds_load_b128 per lane.
// Partials merged into C with global_atomic_add_f32 (C pre-init with bias).
// ---------------------------------------------------------------------------
__global__ __launch_bounds__(256) void gemm_bf16_wmma(
    const __bf16* __restrict__ A,   // [32, K] row-major bf16
    const float*  __restrict__ W,   // [K, N]  row-major f32
    float* __restrict__ C,          // [32, N]  f32 accumulator (bias pre-init)
    int K, int N, int kstages)
{
    __shared__ __attribute__((aligned(16))) __bf16 sW[2][64 * 64];  // [buf][n][k]

    const int t    = threadIdx.x;
    const int wave = t >> 5;            // 0..7
    const int lane = t & 31;
    const int mt   = wave & 1;          // M tile: rows mt*16 .. mt*16+15
    const int nb   = wave >> 1;         // N sub-band: cols nb*16 .. nb*16+15
    const int nbase = blockIdx.x * 64;
    const int k0    = blockIdx.y * kstages * 64;

    const int row = lane & 15;
    const int hi  = lane >> 4;

    // staging role: fixed column nn; k pairs {it*8+kp, it*8+kp+1}, it=0..7.
    // Pairs are adjacent in the [n][k] LDS layout -> one packed b32 store each.
    const int nn = t & 63;
    const int kp = (t >> 6) * 2;        // 0,2,4,6

    float r[16];
    // prologue: prefetch stage 0 (non-temporal: W is stream-once)
    {
        const float* wp = W + (size_t)(k0 + kp) * N + nbase + nn;
        #pragma unroll
        for (int it = 0; it < 8; ++it) {
            r[2 * it]     = __builtin_nontemporal_load(wp + (size_t)(it * 8) * N);
            r[2 * it + 1] = __builtin_nontemporal_load(wp + (size_t)(it * 8 + 1) * N);
        }
    }
    #pragma unroll
    for (int it = 0; it < 8; ++it) {
        v2bf p;
        p.x = (__bf16)r[2 * it];
        p.y = (__bf16)r[2 * it + 1];
        *(v2bf*)(&sW[0][nn * 64 + it * 8 + kp]) = p;
    }
    __syncthreads();

    v8f acc = {};
    for (int s = 0; s < kstages; ++s) {
        const int buf = s & 1;
        const int kg  = k0 + s * 64;

        // prefetch next stage into registers; overlaps the WMMAs below
        if (s + 1 < kstages) {
            const float* wp = W + (size_t)(kg + 64 + kp) * N + nbase + nn;
            #pragma unroll
            for (int it = 0; it < 8; ++it) {
                r[2 * it]     = __builtin_nontemporal_load(wp + (size_t)(it * 8) * N);
                r[2 * it + 1] = __builtin_nontemporal_load(wp + (size_t)(it * 8 + 1) * N);
            }
        }

        // two 16x16x32 WMMAs covering this 64-wide K stage
        #pragma unroll
        for (int kk = 0; kk < 2; ++kk) {
            // A fragment: row mt*16+row; elems 0..7 K=hi*8.., elems 8..15 K=16+hi*8..
            const __bf16* arow =
                A + (size_t)(mt * 16 + row) * K + kg + kk * 32 + hi * 8;
            v8bf a0 = *(const v8bf*)(arow);
            v8bf a1 = *(const v8bf*)(arow + 16);
            v16bf av = __builtin_shufflevector(a0, a1, 0, 1, 2, 3, 4, 5, 6, 7,
                                                       8, 9, 10, 11, 12, 13, 14, 15);
            // B fragment: column n = nb*16+row; elems 0..7 K=kk*32+hi*8.., elems 8..15 K=kk*32+16+hi*8..
            const __bf16* bcol = &sW[buf][(nb * 16 + row) * 64 + kk * 32 + hi * 8];
            v8bf b0 = *(const v8bf*)(bcol);
            v8bf b1 = *(const v8bf*)(bcol + 16);
            v16bf bv = __builtin_shufflevector(b0, b1, 0, 1, 2, 3, 4, 5, 6, 7,
                                                       8, 9, 10, 11, 12, 13, 14, 15);
            acc = __builtin_amdgcn_wmma_f32_16x16x32_bf16(
                false, av, false, bv, (short)0, acc, false, false);
        }

        // stage s+1 into the alternate buffer; reads of it start after the barrier
        if (s + 1 < kstages) {
            #pragma unroll
            for (int it = 0; it < 8; ++it) {
                v2bf p;
                p.x = (__bf16)r[2 * it];
                p.y = (__bf16)r[2 * it + 1];
                *(v2bf*)(&sW[buf ^ 1][nn * 64 + it * 8 + kp]) = p;
            }
        }
        __syncthreads();
    }

    // C layout: VGPR i -> row hi*8+i, col = lane&15
    const int ccol = nbase + nb * 16 + row;
    #pragma unroll
    for (int i = 0; i < 8; ++i) {
        int crow = mt * 16 + hi * 8 + i;
        atomicAdd(&C[(size_t)crow * N + ccol], acc[i]);
    }
}

// ---------------------------------------------------------------------------
extern "C" void kernel_launch(void* const* d_in, const int* in_sizes, int n_in,
                              void* d_out, int out_size, void* d_ws, size_t ws_size,
                              hipStream_t stream)
{
    const float* x    = (const float*)d_in[0];
    const float* mem  = (const float*)d_in[1];
    const float* mtim = (const float*)d_in[2];
    const float* msur = (const float*)d_in[3];
    // d_in[4] last_prediction: only feeds the (unused-downstream) surprise value
    const float* W0   = (const float*)d_in[5];
    const float* b0   = (const float*)d_in[6];
    const float* W1   = (const float*)d_in[7];
    const float* b1   = (const float*)d_in[8];
    const float* W2   = (const float*)d_in[9];
    const float* b2   = (const float*)d_in[10];
    const float* Wout = (const float*)d_in[11];
    const float* bout = (const float*)d_in[12];
    float* out = (float*)d_out;

    char*  ws  = (char*)d_ws;
    size_t off = 0;
    __bf16* A0  = (__bf16*)(ws + off); off += ((size_t)BATCH * DIN * 2 + 255) & ~(size_t)255;
    float*  h0f = (float*) (ws + off); off += (size_t)BATCH * HID * 4;
    __bf16* h0b = (__bf16*)(ws + off); off += (size_t)BATCH * HID * 2;
    float*  h1f = (float*) (ws + off); off += (size_t)BATCH * HID * 4;
    __bf16* h1b = (__bf16*)(ws + off); off += (size_t)BATCH * HID * 2;
    float*  h2f = (float*) (ws + off); off += (size_t)BATCH * HID * 4;
    __bf16* h2b = (__bf16*)(ws + off); off += (size_t)BATCH * HID * 2;
    (void)ws_size; (void)in_sizes; (void)n_in; (void)out_size;

    // 1) build permuted bf16 A-matrix
    prep_kernel<<<BATCH, 256, 0, stream>>>(x, mem, mtim, msur, A0);

    const int nelems = BATCH * HID;

    // 2) layer 0: [32,136192] x [136192,1024]. 2128 K-stages = 38 splits * 56.
    bias_init<<<(nelems + 255) / 256, 256, 0, stream>>>(h0f, b0, HID);
    gemm_bf16_wmma<<<dim3(HID / 64, 38), 256, 0, stream>>>(A0, W0, h0f, DIN, HID, 56);
    relu_to_bf16<<<(nelems + 255) / 256, 256, 0, stream>>>(h0f, h0b, nelems);

    // 3) layer 1: 1024x1024. 16 K-stages = 4 splits * 4.
    bias_init<<<(nelems + 255) / 256, 256, 0, stream>>>(h1f, b1, HID);
    gemm_bf16_wmma<<<dim3(HID / 64, 4), 256, 0, stream>>>(h0b, W1, h1f, HID, HID, 4);
    relu_to_bf16<<<(nelems + 255) / 256, 256, 0, stream>>>(h1f, h1b, nelems);

    // 4) layer 2
    bias_init<<<(nelems + 255) / 256, 256, 0, stream>>>(h2f, b2, HID);
    gemm_bf16_wmma<<<dim3(HID / 64, 4), 256, 0, stream>>>(h1b, W2, h2f, HID, HID, 4);
    relu_to_bf16<<<(nelems + 255) / 256, 256, 0, stream>>>(h2f, h2b, nelems);

    // 5) output layer: [32,1024] x [1024,256] straight into d_out (bias-pre-init,
    //    fully rewritten every call -> deterministic across graph replays)
    bias_init<<<(BATCH * VOC + 255) / 256, 256, 0, stream>>>(out, bout, VOC);
    gemm_bf16_wmma<<<dim3(VOC / 64, 8), 256, 0, stream>>>(h2b, Wout, out, HID, VOC, 2);
}
